// FrequencyAwareHierarchicalEmbedding_29480655520272
// MI455X (gfx1250) — matrix operand order repaired and checked by reading, not compiled, not run on hardware
//
#include <hip/hip_runtime.h>

typedef __attribute__((ext_vector_type(16))) _Float16 v16h;
typedef __attribute__((ext_vector_type(8)))  _Float16 v8h;
typedef __attribute__((ext_vector_type(4)))  _Float16 v4h;
typedef __attribute__((ext_vector_type(8)))  float    v8f;

#define KDIM   257   // 2*D + 1
#define KPAD   288   // padded to 9 chunks of 32 (WMMA K)
#define CSTR   296   // combined row stride in halfs (16B-aligned, bank-staggered)
#define WAVES  4     // waves per block (wave32)
#define TOK_PER_WAVE 16

__global__ __launch_bounds__(WAVES * 32)
void fahe_fused_kernel(const int* __restrict__ fine_ids,
                       const int* __restrict__ coarse_ids,
                       const float* __restrict__ fine_W,
                       const float* __restrict__ coarse_W,
                       const float* __restrict__ freq_W,
                       const float* __restrict__ W1,
                       const float* __restrict__ b1,
                       const float* __restrict__ W2,
                       const float* __restrict__ b2,
                       float* __restrict__ out_fused,
                       float* __restrict__ out_gate,
                       int n_tokens)
{
    // W1^T in f16, zero-padded along K to KPAD: row n holds W1[:,n]
    __shared__ __align__(32) _Float16 sW1t[32 * KPAD];
    // per-wave combined rows: [fine(128) | coarse(128) | freq(1) | zeros..287]
    __shared__ __align__(32) _Float16 sComb[WAVES][TOK_PER_WAVE * CSTR];
    __shared__ float sAg[WAVES][TOK_PER_WAVE];

    const int tid  = threadIdx.x;
    const int wv   = tid >> 5;
    const int ln   = tid & 31;
    const int half = ln >> 4;     // 0: lanes 0-15, 1: lanes 16-31
    const int nlo  = ln & 15;

    // ---- stage W1^T (f16, K-padded) once per block ----
    for (int idx = tid; idx < 32 * KPAD; idx += WAVES * 32) {
        const int n = idx / KPAD;
        const int k = idx - n * KPAD;
        sW1t[idx] = (k < KDIM) ? (_Float16)W1[k * 32 + n] : (_Float16)0.0f;
    }

    const int  base  = (blockIdx.x * WAVES + wv) * TOK_PER_WAVE;
    const bool valid = (base < n_tokens);   // wave-uniform: EXEC stays all-1s

    // ---- gather embeddings, build f16 combined rows in LDS ----
    int myid = 0;
    if (valid) {
        const int tok = base + nlo;
        if (tok < n_tokens)
            myid = (ln < 16) ? fine_ids[tok] : coarse_ids[tok];
    }
    if (valid) {
        _Float16* comb = sComb[wv];
        const int tcnt = min(TOK_PER_WAVE, n_tokens - base);
        for (int t = 0; t < tcnt; ++t) {
            const int fid = __shfl(myid, t, 32);
            const int cid = __shfl(myid, 16 + t, 32);
            const float4 f4 = *(const float4*)(fine_W   + (size_t)fid * 128 + ln * 4);
            const float4 c4 = *(const float4*)(coarse_W + (size_t)cid * 128 + ln * 4);
            v4h fh = { (_Float16)f4.x, (_Float16)f4.y, (_Float16)f4.z, (_Float16)f4.w };
            v4h ch = { (_Float16)c4.x, (_Float16)c4.y, (_Float16)c4.z, (_Float16)c4.w };
            *(v4h*)&comb[t * CSTR + ln * 4]       = fh;
            *(v4h*)&comb[t * CSTR + 128 + ln * 4] = ch;
            if (ln < 31) comb[t * CSTR + KDIM + ln] = (_Float16)0.0f; // zero K=257..287
        }
        if (ln < 16 && nlo < tcnt) {
            const float fr = freq_W[(size_t)myid];              // 1 column
            const float s  = 1.0f / (1.0f + __expf(-fr));        // sigmoid(freq)
            comb[nlo * CSTR + 256] = (_Float16)s;
        }
    }
    __syncthreads();

    // ---- gate MLP layer 1 via WMMA: [16 x 288] @ [288 x 32] ----
    v8f acc0 = {};   // hidden units 0..15
    v8f acc1 = {};   // hidden units 16..31
    if (valid) {
        const _Float16* arow = sComb[wv] + nlo * CSTR;   // A: M = lane&15
        #pragma unroll
        for (int c = 0; c < KPAD / 32; ++c) {
            // A fragment: K = c*32 + 16*half + 0..15 (contiguous)
            const v8h alo = *(const v8h*)(arow + c * 32 + 16 * half);
            const v8h ahi = *(const v8h*)(arow + c * 32 + 16 * half + 8);
            v16h a;
            #pragma unroll
            for (int j = 0; j < 8; ++j) { a[j] = alo[j]; a[8 + j] = ahi[j]; }
            // B fragments: N = lane&15 (+16), K = c*32 + 16*half + 0..15 (contiguous)
            const v16h bf0 = *(const v16h*)&sW1t[nlo        * KPAD + c * 32 + 16 * half];
            const v16h bf1 = *(const v16h*)&sW1t[(nlo + 16) * KPAD + c * 32 + 16 * half];
            acc0 = __builtin_amdgcn_wmma_f32_16x16x32_f16(false, a, false, bf0,
                                                          (short)0, acc0, false, false);
            acc1 = __builtin_amdgcn_wmma_f32_16x16x32_f16(false, a, false, bf1,
                                                          (short)0, acc1, false, false);
        }
    }

    // ---- bias + ReLU + layer 2 dot + sigmoid -> adjusted gate ----
    if (valid) {
        const _Float16* comb = sComb[wv];
        const float b1a = b1[nlo], b1b = b1[nlo + 16];
        const float w2a = W2[nlo], w2b = W2[nlo + 16];
        const float bb2 = b2[0];
        #pragma unroll
        for (int r = 0; r < 8; ++r) {
            // C layout: VGPR r -> token M = r + 8*half, N = lane&15
            const float h0 = fmaxf(acc0[r] + b1a, 0.0f);
            const float h1 = fmaxf(acc1[r] + b1b, 0.0f);
            float p = h0 * w2a + h1 * w2b;
            p += __shfl_xor(p, 1, 16);
            p += __shfl_xor(p, 2, 16);
            p += __shfl_xor(p, 4, 16);
            p += __shfl_xor(p, 8, 16);
            if (nlo == 0) {
                const int t    = r + 8 * half;
                const float g  = 1.0f / (1.0f + __expf(-(p + bb2)));
                const float fq = (float)comb[t * CSTR + 256];
                const float ag = g * fq;
                sAg[wv][t] = ag;
                if (base + t < n_tokens) out_gate[base + t] = ag;
            }
        }
    }
    __syncthreads();

    // ---- gated blend, coalesced 512B stores per token ----
    if (valid) {
        const _Float16* comb = sComb[wv];
        const int tcnt = min(TOK_PER_WAVE, n_tokens - base);
        for (int t = 0; t < tcnt; ++t) {
            const float ag = sAg[wv][t];
            const float om = 1.0f - ag;
            const v4h fh = *(const v4h*)&comb[t * CSTR + ln * 4];
            const v4h ch = *(const v4h*)&comb[t * CSTR + 128 + ln * 4];
            float4 o;
            o.x = ag * (float)fh[0] + om * (float)ch[0];
            o.y = ag * (float)fh[1] + om * (float)ch[1];
            o.z = ag * (float)fh[2] + om * (float)ch[2];
            o.w = ag * (float)fh[3] + om * (float)ch[3];
            *(float4*)(out_fused + (size_t)(base + t) * 128 + ln * 4) = o;
        }
    }
}

extern "C" void kernel_launch(void* const* d_in, const int* in_sizes, int n_in,
                              void* d_out, int out_size, void* d_ws, size_t ws_size,
                              hipStream_t stream) {
    const int*   fine_ids   = (const int*)d_in[0];
    const int*   coarse_ids = (const int*)d_in[1];
    const float* fine_W     = (const float*)d_in[2];
    const float* coarse_W   = (const float*)d_in[3];
    const float* freq_W     = (const float*)d_in[4];
    const float* W1         = (const float*)d_in[5];
    const float* b1         = (const float*)d_in[6];
    const float* W2         = (const float*)d_in[7];
    const float* b2         = (const float*)d_in[8];

    const int n_tokens = in_sizes[0];                 // B*L
    float* out_fused = (float*)d_out;                 // [B*L, 128]
    float* out_gate  = (float*)d_out + (size_t)n_tokens * 128;  // [B*L]

    const int tiles_per_block = WAVES * TOK_PER_WAVE; // 64 tokens / block
    const int blocks = (n_tokens + tiles_per_block - 1) / tiles_per_block;
    fahe_fused_kernel<<<blocks, WAVES * 32, 0, stream>>>(
        fine_ids, coarse_ids, fine_W, coarse_W, freq_W,
        W1, b1, W2, b2, out_fused, out_gate, n_tokens);
}
